// MinkowskiContextualAttention_66185446031533
// MI455X (gfx1250) — matrix operand order repaired and checked by reading, not compiled, not run on hardware
//
#include <hip/hip_runtime.h>
#include <hip/hip_bf16.h>

#define B_    8
#define NF_   2048
#define NK_   4096
#define CIN_  256
#define CKQ_  64
#define COUT_ 256
#define NTOK_ (NF_ + NK_)

typedef __bf16 bf16_t;
typedef __attribute__((ext_vector_type(16))) __bf16 v16bf;
typedef __attribute__((ext_vector_type(8)))  __bf16 v8bf;
typedef __attribute__((ext_vector_type(8)))  float  v8f;
typedef __attribute__((ext_vector_type(4)))  unsigned int v4u;
typedef __attribute__((ext_vector_type(8)))  int v8i_t;
typedef __attribute__((ext_vector_type(4)))  int v4i_t;

static __device__ __forceinline__ v8f wmma_bf16(v16bf a, v16bf b, v8f c) {
  return __builtin_amdgcn_wmma_f32_16x16x32_bf16(false, a, false, b, (short)0, c,
                                                 false, false);
}

// Build a 16-element B/A fragment from two 16-byte chunks (works for LDS and
// global pointers that are only 16B-aligned).
static __device__ __forceinline__ v16bf ld_frag2(const bf16_t* p) {
  v8bf lo = *(const v8bf*)p;
  v8bf hi = *(const v8bf*)(p + 8);
  v16bf r;
#pragma unroll
  for (int e = 0; e < 8; ++e) { r[e] = lo[e]; r[8 + e] = hi[e]; }
  return r;
}

// ---------------------------------------------------------------------------
// Tensor Data Mover: async 2-D tile load global -> LDS with row padding.
// D# layout per CDNA5 ISA (groups 0/1; 2-3 zero).
// Dims/strides in data_size (=2B) units. pad codes: interval 2^(pi+1) DWORDs,
// amount (pa+1) DWORDs. Toolchain exposes the 6-arg builtin form.
// ---------------------------------------------------------------------------
static __device__ __forceinline__ void tdm_load_2d(
    unsigned lds_off, const bf16_t* gptr,
    unsigned tensor_d0, unsigned tensor_d1, unsigned long long d0_stride,
    unsigned tile_d0, unsigned tile_d1, unsigned pi, unsigned pa) {
  unsigned long long ga = (unsigned long long)(size_t)gptr;
  v4u g0;
  g0[0] = 1u;                                  // count=1, user descriptor
  g0[1] = lds_off;                             // LDS byte address
  g0[2] = (unsigned)ga;                        // global addr [31:0]
  g0[3] = (unsigned)(ga >> 32) | (2u << 30);   // global addr [56:32] | type=2
  v8i_t g1;
  g1[0] = (int)((1u << 16)        // data_size = 2 bytes
              | (1u << 20)        // pad_enable
              | (pi << 22)        // pad_interval code
              | (pa << 25));      // pad_amount code
  g1[1] = (int)(tensor_d0 << 16);                              // dim0 [15:0]
  g1[2] = (int)((tensor_d0 >> 16) | (tensor_d1 << 16));        // dim0 hi, dim1 lo
  g1[3] = (int)((tensor_d1 >> 16) | (tile_d0 << 16));          // dim1 hi, tile0
  g1[4] = (int)(tile_d1 & 0xFFFFu);                            // tile1 (tile2=0)
  g1[5] = (int)(unsigned)(d0_stride & 0xFFFFFFFFull);          // stride0 lo
  g1[6] = (int)(unsigned)((d0_stride >> 32) & 0xFFFFull);      // stride0 hi
  g1[7] = 0;
  v4i_t gz4 = {0, 0, 0, 0};
  v8i_t gz8 = {0, 0, 0, 0, 0, 0, 0, 0};
  __builtin_amdgcn_tensor_load_to_lds(g0, g1, gz4, gz4, gz8, 0);
}

// ---------------------------------------------------------------------------
// Converters
// ---------------------------------------------------------------------------
__global__ __launch_bounds__(256)
void conv_feat_kernel(const float* __restrict__ f, bf16_t* __restrict__ fb,
                      float* __restrict__ out) {
  size_t i = (size_t)blockIdx.x * 256 + threadIdx.x;
  float v = f[i];
  fb[i] = (bf16_t)v;
  int tok = (int)((i >> 8) % NTOK_);
  if (tok >= NF_) out[i] = v;  // keep rows pass straight through
}

__global__ __launch_bounds__(256)
void conv_w_kernel(const float* __restrict__ s, bf16_t* __restrict__ d, int n) {
  int i = blockIdx.x * 256 + threadIdx.x;
  if (i < n) d[i] = (bf16_t)s[i];
}

// ---------------------------------------------------------------------------
// WMMA projection: out(rows x NOUT) = F(rows x 256) * W^T + bias, times scale.
// ---------------------------------------------------------------------------
template <int NOUT, bool TRANS>
__global__ __launch_bounds__(128)
void proj_wmma_kernel(const bf16_t* __restrict__ Fbf, const bf16_t* __restrict__ Wbf,
                      const float* __restrict__ bias, bf16_t* __restrict__ dst,
                      int rows_per_b, int row_off, float scale) {
  const int wave = threadIdx.x >> 5;
  const int lane = threadIdx.x & 31;
  const int hf   = lane >> 4;
  const int l16  = lane & 15;
  const int arow = l16;
  const int kb   = hf * 16;

  const int tiles_pb = rows_per_b / 16;
  const int tile = blockIdx.x * 4 + wave;
  const int b    = tile / tiles_pb;
  const int r0   = (tile - b * tiles_pb) * 16;

  const bf16_t* F = Fbf + ((size_t)b * NTOK_ + row_off + r0) * CIN_;

  constexpr int CT = NOUT / 16;
  v8f zero = {};
  v8f acc[CT];
#pragma unroll
  for (int t = 0; t < CT; ++t) acc[t] = zero;

#pragma unroll
  for (int ks = 0; ks < CIN_ / 32; ++ks) {
    v16bf a;
    {
      const bf16_t* fr = F + (size_t)arow * CIN_ + ks * 32 + kb;
      v8bf lo = *(const v8bf*)(fr);
      v8bf hi = *(const v8bf*)(fr + 8);
#pragma unroll
      for (int e = 0; e < 8; ++e) { a[e] = lo[e]; a[8 + e] = hi[e]; }
    }
#pragma unroll
    for (int ct = 0; ct < CT; ++ct) {
      v16bf wb = *(const v16bf*)(Wbf + (size_t)(ct * 16 + l16) * CIN_ + ks * 32 + hf * 16);
      acc[ct] = wmma_bf16(a, wb, acc[ct]);
    }
  }

#pragma unroll
  for (int ct = 0; ct < CT; ++ct) {
    int col = ct * 16 + l16;
    if (TRANS) {
      // 8 consecutive rows of Vt -> one 16B store per (ct, lane)
      v8bf pk;
#pragma unroll
      for (int j = 0; j < 8; ++j)
        pk[j] = (bf16_t)((acc[ct][j] + bias[col]) * scale);
      *(v8bf*)(dst + ((size_t)b * NOUT + col) * NK_ + r0 + hf * 8) = pk;
    } else {
#pragma unroll
      for (int j = 0; j < 8; ++j) {
        int row = hf * 8 + j;
        float v = (acc[ct][j] + bias[col]) * scale;
        dst[((size_t)(b * rows_per_b + r0 + row)) * NOUT + col] = (bf16_t)v;
      }
    }
  }
}

// ---------------------------------------------------------------------------
// Two-pass flash attention with TDM double-buffered K/Vt staging in pass 2.
// grid = B_ * (NF_/64) = 256 blocks, 128 threads (4 waves x 16 Q rows).
// ---------------------------------------------------------------------------
__global__ __launch_bounds__(128)
void attn_kernel(const bf16_t* __restrict__ Qw, const bf16_t* __restrict__ Kw,
                 const bf16_t* __restrict__ Vtw, float* __restrict__ out) {
  // Vt chunk: 256 rows x 32 elems, padded stride 40 elems (80B): banks 20*r%64
  // K  chunk:  32 rows x 64 elems, padded stride 72 elems (144B): banks 36*r%64
  __shared__ bf16_t vbuf[2][COUT_ * 40];
  __shared__ bf16_t kbuf[2][32 * 72];
  __shared__ bf16_t pbuf[4][16 * 40];

  const int wave = threadIdx.x >> 5;
  const int lane = threadIdx.x & 31;
  const int hf   = lane >> 4;
  const int l16  = lane & 15;
  const int arow = l16;
  const int kb   = hf * 16;

  const int blk   = blockIdx.x;
  const int b     = blk >> 5;
  const int qblk  = blk & 31;
  const int qrow0 = qblk * 64 + wave * 16;

  const bf16_t* Q  = Qw  + ((size_t)b * NF_ + qrow0) * CKQ_;
  const bf16_t* Kp = Kw  + (size_t)b * NK_ * CKQ_;
  const bf16_t* Vt = Vtw + (size_t)b * COUT_ * NK_;
  bf16_t* pb = pbuf[wave];

  // Q tile as two A-fragments (K 0..31, 32..63)
  v16bf qa[2];
#pragma unroll
  for (int s = 0; s < 2; ++s) {
    const bf16_t* qr = Q + (size_t)arow * CKQ_ + s * 32 + kb;
    v8bf lo = *(const v8bf*)(qr);
    v8bf hi = *(const v8bf*)(qr + 8);
#pragma unroll
    for (int e = 0; e < 8; ++e) { qa[s][e] = lo[e]; qa[s][8 + e] = hi[e]; }
  }

  v8f zero = {};

  // ---------------- pass 1: row max (K from global/L2) ----------------
  float rM[8];
#pragma unroll
  for (int j = 0; j < 8; ++j) rM[j] = -3.0e38f;

  for (int kv0 = 0; kv0 < NK_; kv0 += 32) {
    if (kv0 + 32 < NK_)
      __builtin_prefetch(Kp + (size_t)(kv0 + 32 + l16) * CKQ_, 0, 0);
    v8f s0 = zero, s1 = zero;
#pragma unroll
    for (int s = 0; s < 2; ++s) {
      v16bf b0 = *(const v16bf*)(Kp + (size_t)(kv0 + l16) * CKQ_ + s * 32 + hf * 16);
      v16bf b1 = *(const v16bf*)(Kp + (size_t)(kv0 + 16 + l16) * CKQ_ + s * 32 + hf * 16);
      s0 = wmma_bf16(qa[s], b0, s0);
      s1 = wmma_bf16(qa[s], b1, s1);
    }
#pragma unroll
    for (int j = 0; j < 8; ++j) rM[j] = fmaxf(rM[j], fmaxf(s0[j], s1[j]));
  }
#pragma unroll
  for (int j = 0; j < 8; ++j) {
    float m = rM[j];
    m = fmaxf(m, __shfl_xor(m, 1));
    m = fmaxf(m, __shfl_xor(m, 2));
    m = fmaxf(m, __shfl_xor(m, 4));
    m = fmaxf(m, __shfl_xor(m, 8));
    rM[j] = m;
  }

  // ---------------- pass 2: exp + P*V (K/Vt staged by TDM) ----------------
  v8f acc[16];
#pragma unroll
  for (int t = 0; t < 16; ++t) acc[t] = zero;
  float rS[8];
#pragma unroll
  for (int j = 0; j < 8; ++j) rS[j] = 0.f;

  const int NCH = NK_ / 32;
  if (wave == 0) {
    tdm_load_2d((unsigned)(size_t)&kbuf[0][0], Kp,
                /*t_d0*/CKQ_, /*t_d1*/NK_, /*stride*/CKQ_,
                /*tile0*/CKQ_, /*tile1*/32, /*pi*/4, /*pa*/3);
    tdm_load_2d((unsigned)(size_t)&vbuf[0][0], Vt,
                /*t_d0*/NK_, /*t_d1*/COUT_, /*stride*/NK_,
                /*tile0*/32, /*tile1*/COUT_, /*pi*/3, /*pa*/3);
  }

  for (int ch = 0; ch < NCH; ++ch) {
    const int cur = ch & 1;
    if (wave == 0) __builtin_amdgcn_s_wait_tensorcnt(0);
    __syncthreads();  // buffer `cur` ready for everyone
    if (wave == 0 && ch + 1 < NCH) {
      const int nk = (ch + 1) * 32;
      tdm_load_2d((unsigned)(size_t)&kbuf[cur ^ 1][0], Kp + (size_t)nk * CKQ_,
                  CKQ_, NK_, CKQ_, CKQ_, 32, 4, 3);
      tdm_load_2d((unsigned)(size_t)&vbuf[cur ^ 1][0], Vt + nk,
                  NK_, COUT_, NK_, 32, COUT_, 3, 3);
    }
    const bf16_t* kl = kbuf[cur];
    const bf16_t* vl = vbuf[cur];

    v8f s0 = zero, s1 = zero;
#pragma unroll
    for (int s = 0; s < 2; ++s) {
      v16bf b0 = ld_frag2(kl + (size_t)l16 * 72 + s * 32 + hf * 16);
      v16bf b1 = ld_frag2(kl + (size_t)(16 + l16) * 72 + s * 32 + hf * 16);
      s0 = wmma_bf16(qa[s], b0, s0);
      s1 = wmma_bf16(qa[s], b1, s1);
    }
#pragma unroll
    for (int j = 0; j < 8; ++j) {
      float p0 = __expf(s0[j] - rM[j]);
      float p1 = __expf(s1[j] - rM[j]);
      rS[j] += p0 + p1;
      s0[j] = p0; s1[j] = p1;
    }
    // C-layout -> A-layout via per-wave LDS tile
#pragma unroll
    for (int j = 0; j < 8; ++j) {
      int row = hf * 8 + j;
      pb[row * 40 + l16]      = (bf16_t)s0[j];
      pb[row * 40 + 16 + l16] = (bf16_t)s1[j];
    }
    v16bf pa;
    {
      v8bf lo = *(const v8bf*)(pb + arow * 40 + kb);
      v8bf hi = *(const v8bf*)(pb + arow * 40 + 8 + kb);
#pragma unroll
      for (int e = 0; e < 8; ++e) { pa[e] = lo[e]; pa[8 + e] = hi[e]; }
    }
#pragma unroll
    for (int t = 0; t < 16; ++t) {
      v16bf vb = ld_frag2(vl + (size_t)(t * 16 + l16) * 40 + hf * 16);
      acc[t] = wmma_bf16(pa, vb, acc[t]);
    }
    __syncthreads();  // everyone done reading buffer `cur`
  }

  // final row-sum butterfly, normalize, store
  float* orow = out + ((size_t)b * NTOK_ + qrow0) * COUT_;
#pragma unroll
  for (int j = 0; j < 8; ++j) {
    float s = rS[j];
    s += __shfl_xor(s, 1);
    s += __shfl_xor(s, 2);
    s += __shfl_xor(s, 4);
    s += __shfl_xor(s, 8);
    float inv = 1.0f / s;
    int row = hf * 8 + j;
#pragma unroll
    for (int t = 0; t < 16; ++t)
      orow[(size_t)row * COUT_ + t * 16 + l16] = acc[t][j] * inv;
  }
}

// ---------------------------------------------------------------------------
extern "C" void kernel_launch(void* const* d_in, const int* in_sizes, int n_in,
                              void* d_out, int out_size, void* d_ws, size_t ws_size,
                              hipStream_t stream) {
  const float* features = (const float*)d_in[0];
  const float* Wq = (const float*)d_in[2];
  const float* bq = (const float*)d_in[3];
  const float* Wk = (const float*)d_in[4];
  const float* bk = (const float*)d_in[5];
  const float* Wv = (const float*)d_in[6];
  const float* bv = (const float*)d_in[7];
  float* out = (float*)d_out;

  bf16_t* Fbf  = (bf16_t*)d_ws;
  bf16_t* Qws  = Fbf  + (size_t)B_ * NTOK_ * CIN_;
  bf16_t* Kws  = Qws  + (size_t)B_ * NF_ * CKQ_;
  bf16_t* Vtws = Kws  + (size_t)B_ * NK_ * CKQ_;
  bf16_t* Wqb  = Vtws + (size_t)B_ * COUT_ * NK_;
  bf16_t* Wkb  = Wqb + CKQ_ * CIN_;
  bf16_t* Wvb  = Wkb + CKQ_ * CIN_;

  const float inv_sqrt = 0.125f;  // 1/sqrt(64)

  conv_feat_kernel<<<(B_ * NTOK_ * CIN_) / 256, 256, 0, stream>>>(features, Fbf, out);
  conv_w_kernel<<<(CKQ_ * CIN_ + 255) / 256, 256, 0, stream>>>(Wq, Wqb, CKQ_ * CIN_);
  conv_w_kernel<<<(CKQ_ * CIN_ + 255) / 256, 256, 0, stream>>>(Wk, Wkb, CKQ_ * CIN_);
  conv_w_kernel<<<(COUT_ * CIN_ + 255) / 256, 256, 0, stream>>>(Wv, Wvb, COUT_ * CIN_);

  proj_wmma_kernel<CKQ_, false><<<B_ * NF_ / 64, 128, 0, stream>>>(
      Fbf, Wqb, bq, Qws, NF_, 0, inv_sqrt);
  proj_wmma_kernel<CKQ_, false><<<B_ * NK_ / 64, 128, 0, stream>>>(
      Fbf, Wkb, bk, Kws, NK_, NF_, 1.0f);
  proj_wmma_kernel<COUT_, true><<<B_ * NK_ / 64, 128, 0, stream>>>(
      Fbf, Wvb, bv, Vtws, NK_, NF_, 1.0f);

  attn_kernel<<<B_ * (NF_ / 64), 128, 0, stream>>>(Qws, Kws, Vtws, out);
}
